// Correlation_TRT_23252952940995
// MI455X (gfx1250) — matrix operand order repaired and checked by reading, not compiled, not run on hardware
//
#include <hip/hip_runtime.h>

// Correlation cost volume: out[b,d,h,w] = mean_c( L[b,c,h,w] * R[b,c,h,w+d-40] )
// B=4 C=256 H=128 W=416, D=81. Banded Gram matrix per (b,h) row via
// V_WMMA_F32_16X16X32_F16 with f16 hi/lo split (hh+hl+lh) for ~fp32 accuracy.
// NT=128 w-slice per block, 8 waves x 6 band tiles, K chunked by 64 in LDS.

#define MAXD   40
#define NDISP  81
#define BB     4
#define CC     256
#define HH     128
#define WW     416

#define NT     128             // output w columns per workgroup
#define MR     (NT + 2*MAXD)   // 208 right rows staged (halo)
#define NSLICE 4               // ceil(416/128)
#define CHUNK  64              // K chunk staged in LDS
#define NCHUNK (CC / CHUNK)    // 4
#define SROW   72              // LDS row stride in halves (CHUNK + 8 -> bank rotation)

typedef __attribute__((ext_vector_type(16))) _Float16 v16h;
typedef __attribute__((ext_vector_type(8)))  _Float16 v8h;
typedef __attribute__((ext_vector_type(8)))  float    v8f;
typedef __attribute__((ext_vector_type(4)))  float    f32x4;

// LDS layout (halves). R planes first so all A-fragment ds offsets (<59.9KB)
// and, via a second base pointer, all B-fragment offsets fit 16-bit immediates.
#define R_HI 0
#define R_LO (MR * SROW)
#define L_OFF (2 * MR * SROW)      // L region base
#define L_HI 0                     // relative to smemB
#define L_LO (NT * SROW)           // relative to smemB
#define SMEM_HALVES (2 * (NT + MR) * SROW)   // 48384 halves = 96768 B

union H2 { unsigned u; _Float16 h[2]; };
union F16x16 { v16h v; v8h h[2]; };

__device__ __forceinline__ void cvt_pair(float x0, float x1, unsigned& uh, unsigned& ul) {
  const _Float16 h0 = (_Float16)x0, h1 = (_Float16)x1;
  const _Float16 l0 = (_Float16)(x0 - (float)h0);
  const _Float16 l1 = (_Float16)(x1 - (float)h1);
  H2 a; a.h[0] = h0; a.h[1] = h1; uh = a.u;
  H2 b; b.h[0] = l0; b.h[1] = l1; ul = b.u;
}

__global__ __launch_bounds__(256)
void Correlation_TRT_23252952940995_kernel(const float* __restrict__ left,
                                           const float* __restrict__ right,
                                           float* __restrict__ out) {
  extern __shared__ _Float16 smem[];
  _Float16* __restrict__ smemR = smem;          // right planes (A fragments)
  _Float16* __restrict__ smemB = smem + L_OFF;  // left planes  (B fragments)

  const int tid  = threadIdx.x;
  const int lane = tid & 31;
  const int jj   = tid >> 5;        // wave id = N-tile index, 0..7

  const int bid  = blockIdx.x;
  const int ws   = bid % NSLICE;
  const int hrow = (bid / NSLICE) % HH;
  const int b    = bid / (NSLICE * HH);
  const int w0   = ws * NT;

  const int cstride = HH * WW;   // 53248 floats between channels
  const float* Lg = left  + (size_t)b * CC * cstride + (size_t)hrow * WW;
  const float* Rg = right + (size_t)b * CC * cstride + (size_t)hrow * WW;

  const v8f zero = {0.f,0.f,0.f,0.f,0.f,0.f,0.f,0.f};
  v8f acc[6];
  #pragma unroll
  for (int t = 0; t < 6; ++t) acc[t] = zero;

  const int ml = lane & 15;   // A row / B,C,D column within tile
  const int hf = lane >> 4;   // lane half

  for (int ck = 0; ck < NCHUNK; ++ck) {
    const int cbase = ck * CHUNK;

    // ---- stage LEFT [NT x CHUNK]: float4 along w, channel pairs -> b32 LDS ----
    for (int p = tid; p < (NT / 4) * (CHUNK / 2); p += 256) {   // 1024
      const int w4 = p & 31;           // consecutive tid -> consecutive w (coalesced)
      const int c2 = p >> 5;
      const int wb = w0 + 4 * w4;
      const float* p0 = Lg + (size_t)(cbase + 2 * c2) * cstride;
      const float* p1 = p0 + cstride;
      f32x4 x0 = {0.f,0.f,0.f,0.f}, x1 = x0;
      if (wb + 3 < WW) {
        x0 = __builtin_nontemporal_load((const f32x4*)(p0 + wb));   // L read exactly once
        x1 = __builtin_nontemporal_load((const f32x4*)(p1 + wb));
      } else {
        #pragma unroll
        for (int e = 0; e < 4; ++e)
          if (wb + e < WW) { x0[e] = p0[wb + e]; x1[e] = p1[wb + e]; }
      }
      #pragma unroll
      for (int e = 0; e < 4; ++e) {
        unsigned uh, ul; cvt_pair(x0[e], x1[e], uh, ul);
        const int off = (4 * w4 + e) * SROW + 2 * c2;
        *reinterpret_cast<unsigned*>(&smemB[L_HI + off]) = uh;
        *reinterpret_cast<unsigned*>(&smemB[L_LO + off]) = ul;
      }
    }
    // ---- stage RIGHT [MR x CHUNK] with +-40 halo, zero outside image ----
    for (int p = tid; p < (MR / 4) * (CHUNK / 2); p += 256) {    // 1664
      const int w4 = p % 52;
      const int c2 = p / 52;
      const int wb = w0 - MAXD + 4 * w4;
      const float* p0 = Rg + (size_t)(cbase + 2 * c2) * cstride;
      const float* p1 = p0 + cstride;
      f32x4 x0 = {0.f,0.f,0.f,0.f}, x1 = x0;
      if (wb >= 0 && wb + 3 < WW) {
        x0 = *(const f32x4*)(p0 + wb);
        x1 = *(const f32x4*)(p1 + wb);
      } else {
        #pragma unroll
        for (int e = 0; e < 4; ++e)
          if (wb + e >= 0 && wb + e < WW) { x0[e] = p0[wb + e]; x1[e] = p1[wb + e]; }
      }
      #pragma unroll
      for (int e = 0; e < 4; ++e) {
        unsigned uh, ul; cvt_pair(x0[e], x1[e], uh, ul);
        const int off = (4 * w4 + e) * SROW + 2 * c2;
        *reinterpret_cast<unsigned*>(&smemR[R_HI + off]) = uh;
        *reinterpret_cast<unsigned*>(&smemR[R_LO + off]) = ul;
      }
    }
    // ---- prefetch next K chunk into cache while this chunk computes ----
    if (ck + 1 < NCHUNK) {
      const int cn = cbase + CHUNK;
      for (int p = tid; p < 384; p += 256) {
        if (p < 128) {
          const int c  = cn + (p >> 1);
          const int wb = w0 + (p & 1) * 64;
          if (wb < WW) __builtin_prefetch(Lg + (size_t)c * cstride + wb, 0, 2);
        } else {
          const int q  = p - 128;
          const int c  = cn + (q >> 2);
          int wb = w0 - MAXD + (q & 3) * 64;
          wb = wb < 0 ? 0 : (wb >= WW ? WW - 1 : wb);
          __builtin_prefetch(Rg + (size_t)c * cstride + wb, 0, 2);
        }
      }
    }
    __syncthreads();

    // ---- WMMA over this K chunk: wave jj owns N-tile jj, band M-tiles jj..jj+5 ----
    #pragma unroll
    for (int kc = 0; kc < CHUNK; kc += 32) {
      // B fragment (LEFT as KxN): lane col n=ml; K = kc + 16*hf .. +15 contiguous
      const int bbase = (16 * jj + ml) * SROW + kc + 16 * hf;
      F16x16 ubh, ubl;
      ubh.h[0] = *reinterpret_cast<const v8h*>(&smemB[L_HI + bbase]);
      ubh.h[1] = *reinterpret_cast<const v8h*>(&smemB[L_HI + bbase + 8]);
      ubl.h[0] = *reinterpret_cast<const v8h*>(&smemB[L_LO + bbase]);
      ubl.h[1] = *reinterpret_cast<const v8h*>(&smemB[L_LO + bbase + 8]);

      #pragma unroll
      for (int t = 0; t < 6; ++t) {
        // A fragment (RIGHT as MxK): lane row m=ml; K = kc + 16*hf .. +15 contiguous
        const int abase = (16 * (jj + t) + ml) * SROW + kc + 16 * hf;
        F16x16 uah, ual;
        uah.h[0] = *reinterpret_cast<const v8h*>(&smemR[R_HI + abase]);
        uah.h[1] = *reinterpret_cast<const v8h*>(&smemR[R_HI + abase + 8]);
        ual.h[0] = *reinterpret_cast<const v8h*>(&smemR[R_LO + abase]);
        ual.h[1] = *reinterpret_cast<const v8h*>(&smemR[R_LO + abase + 8]);

        acc[t] = __builtin_amdgcn_wmma_f32_16x16x32_f16(
            false, uah.v, false, ubh.v, (short)0, acc[t], false, false);   // hi*hi
        acc[t] = __builtin_amdgcn_wmma_f32_16x16x32_f16(
            false, uah.v, false, ubl.v, (short)0, acc[t], false, false);   // hi*lo
        acc[t] = __builtin_amdgcn_wmma_f32_16x16x32_f16(
            false, ual.v, false, ubh.v, (short)0, acc[t], false, false);   // lo*hi
      }
    }
    __syncthreads();
  }

  // ---- epilogue: scatter band elements d = w' - w + 40 = 16*t + m - n ----
  const int n = ml;
  const int w = w0 + 16 * jj + n;
  #pragma unroll
  for (int t = 0; t < 6; ++t) {
    #pragma unroll
    for (int v = 0; v < 8; ++v) {
      const int mm = v + 8 * hf;          // C/D layout: M = vgpr + 8*(lane>=16)
      const int d  = 16 * t + mm - n;
      if (d >= 0 && d < NDISP && w < WW) {
        __builtin_nontemporal_store(
            acc[t][v] * (1.0f / (float)CC),
            &out[(((size_t)b * NDISP + d) * HH + hrow) * WW + w]);
      }
    }
  }
}

extern "C" void kernel_launch(void* const* d_in, const int* in_sizes, int n_in,
                              void* d_out, int out_size, void* d_ws, size_t ws_size,
                              hipStream_t stream) {
  (void)in_sizes; (void)n_in; (void)d_ws; (void)ws_size; (void)out_size;
  const float* left  = (const float*)d_in[0];
  const float* right = (const float*)d_in[1];
  float* out = (float*)d_out;

  dim3 grid(NSLICE * HH * BB);   // 4 * 128 * 4 = 2048 workgroups
  dim3 block(256);               // 8 waves (wave32)
  const size_t shmem = (size_t)SMEM_HALVES * sizeof(_Float16);  // 96768 B
  hipLaunchKernelGGL(Correlation_TRT_23252952940995_kernel, grid, block, shmem, stream,
                     left, right, out);
}
